// GroupQueryAttention_3040836846153
// MI455X (gfx1250) — matrix operand order repaired and checked by reading, not compiled, not run on hardware
//
#include <hip/hip_runtime.h>
#include <math.h>

// ---------------------------------------------------------------------------
// GQA forward for MI455X (gfx1250): bf16 WMMA + Tensor Data Mover staging.
// HIDDEN=4096, S=2048, NUM_HEADS=32, GROUP_NUM=8, HEAD_DIM=128.
// ---------------------------------------------------------------------------

typedef __attribute__((ext_vector_type(16))) __bf16       v16bf;
typedef __attribute__((ext_vector_type(8)))  float        v8f;
typedef __attribute__((ext_vector_type(4)))  unsigned int v4u;
typedef __attribute__((ext_vector_type(8)))  int          v8i;
typedef __attribute__((ext_vector_type(4)))  int          v4i;

#define S_LEN     2048
#define HIDDEN    4096
#define KV_DIM    1024
#define HEAD_DIM  128
#define NUM_HEADS 32

// ---- WMMA helper -----------------------------------------------------------
__device__ __forceinline__ v8f wmma_bf16(v16bf a, v16bf b, v8f c) {
  return __builtin_amdgcn_wmma_f32_16x16x32_bf16(
      false, a, false, b, (short)0, c, false, false);
}

// A-fragment (16x32, 16-bit): lane half hi holds K = hi*16 + {0..15}.
__device__ __forceinline__ v16bf ldfragA(const __bf16* p) {
  union { v16bf v; uint4 u[2]; } t;
  t.u[0] = *(const uint4*)(p);
  t.u[1] = *(const uint4*)(p + 8);
  return t.v;
}
// B-fragment (32x16, 16-bit): lane half hi holds K = hi*16 + {0..15}.
__device__ __forceinline__ v16bf ldfragB(const __bf16* p) {
  union { v16bf v; uint4 u[2]; } t;
  t.u[0] = *(const uint4*)(p);
  t.u[1] = *(const uint4*)(p + 8);
  return t.v;
}

// ---- Tensor Data Mover: 2D bf16 tile (tile_d1 rows of tile_d0 elems) -------
// Global rows are stride_d0 elements apart; LDS receives packed rows.
__device__ __forceinline__ unsigned lds_off(const void* p) {
  return (unsigned)(size_t)p;  // LDS aperture: low 32 bits == LDS byte offset
}

__device__ __forceinline__ void tdm_load_2d(unsigned lds_byte_off,
                                            const void* gptr,
                                            unsigned tile_d0, unsigned tile_d1,
                                            unsigned stride_d0) {
  unsigned long long ga = (unsigned long long)(size_t)gptr;
  v4u g0 = {1u,                                   // count=1, user descriptor
            lds_byte_off,                          // lds_addr
            (unsigned)ga,                          // global_addr[31:0]
            (unsigned)(ga >> 32) | (2u << 30)};    // global_addr[56:32], type=2
  union { v8i v; unsigned u[8]; } g1;
  g1.u[0] = 1u << 16;                              // data_size=1 -> 2 bytes
  g1.u[1] = (tile_d0 & 0xFFFFu) << 16;             // tensor_dim0[15:0]
  g1.u[2] = (tile_d0 >> 16) | ((tile_d1 & 0xFFFFu) << 16);  // dim0 hi | dim1 lo
  g1.u[3] = (tile_d1 >> 16) | ((tile_d0 & 0xFFFFu) << 16);  // dim1 hi | tile_dim0
  g1.u[4] = tile_d1 & 0xFFFFu;                     // tile_dim1 (tile_dim2=0)
  g1.u[5] = stride_d0;                             // tensor_dim0_stride[31:0]
  g1.u[6] = 0;
  g1.u[7] = 0;
  v4i z = (v4i)0;
#if defined(__clang_major__) && __clang_major__ >= 23
  __builtin_amdgcn_tensor_load_to_lds(g0, g1.v, z, z, (v8i)0, 0);
#else
  __builtin_amdgcn_tensor_load_to_lds(g0, g1.v, z, z, 0);
#endif
}

// ---- f32 -> bf16 convert (linear) ------------------------------------------
__global__ void cvt_f32_bf16(const float* __restrict__ in,
                             __bf16* __restrict__ out, int n) {
  int i = blockIdx.x * blockDim.x + threadIdx.x;
  int stride = gridDim.x * blockDim.x;
  for (; i < n; i += stride) out[i] = (__bf16)in[i];
}

// ---- f32 [R][C] -> bf16 transposed [C][R] (tiled) --------------------------
__global__ void __launch_bounds__(256)
transpose_cvt(const float* __restrict__ in, __bf16* __restrict__ out,
              int R, int C) {
  __shared__ __bf16 tile[32][33];
  const int bx = blockIdx.x * 32;  // column base
  const int by = blockIdx.y * 32;  // row base
  const int tx = threadIdx.x & 31;
  const int ty = threadIdx.x >> 5;  // 0..7
#pragma unroll
  for (int i = 0; i < 32; i += 8)
    tile[ty + i][tx] = (__bf16)in[(size_t)(by + ty + i) * C + bx + tx];
  __syncthreads();
#pragma unroll
  for (int i = 0; i < 32; i += 8)
    out[(size_t)(bx + ty + i) * R + by + tx] = tile[tx][ty + i];
}

// ---- Tiled WMMA GEMM with TDM double-buffered staging ----------------------
// C[M,N] = A[M,K] @ BT[N,K]^T + bias.  Block 256 threads = 8 waves.
// BM=128, BN=128, BK=32. Wave (w&3)->32-row band, (w>>2)->64-col band.
template <bool BF16_OUT, bool TRANS_OUT>
__global__ void __launch_bounds__(256)
gemm_bias(const __bf16* __restrict__ A, const __bf16* __restrict__ BT,
          const float* __restrict__ bias, void* __restrict__ Cout,
          int M, int N, int K) {
  __shared__ __align__(16) __bf16 sA[2][128 * 32];  // [row][k]
  __shared__ __align__(16) __bf16 sB[2][128 * 32];  // [col][k]

  const int t    = threadIdx.x;
  const int lane = t & 31;
  const int w    = t >> 5;
  const int l    = lane & 15;
  const int hi   = (lane >> 4) & 1;
  const int m0   = blockIdx.y * 128;
  const int n0   = blockIdx.x * 128;
  const int mw   = (w & 3) * 32;
  const int nw   = (w >> 2) * 64;

  v8f acc[2][4];
#pragma unroll
  for (int rt = 0; rt < 2; ++rt)
#pragma unroll
    for (int ct = 0; ct < 4; ++ct) acc[rt][ct] = (v8f)0.0f;

  const int niter = K / 32;

  // Prime buffer 0 via Tensor Data Mover (wave 0 issues; EXEC-independent).
  if (t < 32) {
    tdm_load_2d(lds_off(sA[0]), A + (size_t)m0 * K, 32, 128, (unsigned)K);
    tdm_load_2d(lds_off(sB[0]), BT + (size_t)n0 * K, 32, 128, (unsigned)K);
  }

  for (int it = 0; it < niter; ++it) {
    const int cur = it & 1;
    if (t < 32) {
      if (it + 1 < niter) {
        const int k0n = (it + 1) * 32;
        tdm_load_2d(lds_off(sA[1 - cur]), A + (size_t)m0 * K + k0n, 32, 128,
                    (unsigned)K);
        tdm_load_2d(lds_off(sB[1 - cur]), BT + (size_t)n0 * K + k0n, 32, 128,
                    (unsigned)K);
        __builtin_amdgcn_s_wait_tensorcnt(2);  // current tile pair complete
      } else {
        __builtin_amdgcn_s_wait_tensorcnt(0);
      }
    }
    __syncthreads();

    const __bf16* a = sA[cur];
    const __bf16* b = sB[cur];
    v16bf af[2], bfr[4];
#pragma unroll
    for (int rt = 0; rt < 2; ++rt)
      af[rt] = ldfragA(a + (mw + rt * 16 + l) * 32 + hi * 16);
#pragma unroll
    for (int ct = 0; ct < 4; ++ct)
      bfr[ct] = ldfragB(b + (nw + ct * 16 + l) * 32 + hi * 16);
#pragma unroll
    for (int rt = 0; rt < 2; ++rt)
#pragma unroll
      for (int ct = 0; ct < 4; ++ct)
        acc[rt][ct] = wmma_bf16(af[rt], bfr[ct], acc[rt][ct]);
    __syncthreads();  // done reading buf[cur] before TDM refills it
  }

  // Epilogue: bias add, store (C layout: VGPR r -> row hi*8+r, col = lane&15)
#pragma unroll
  for (int rt = 0; rt < 2; ++rt) {
#pragma unroll
    for (int ct = 0; ct < 4; ++ct) {
      int col = n0 + nw + ct * 16 + l;
      float bv = bias[col];
#pragma unroll
      for (int r = 0; r < 8; ++r) {
        int row = m0 + mw + rt * 16 + hi * 8 + r;
        float v = acc[rt][ct][r] + bv;
        if (BF16_OUT) {
          if (TRANS_OUT)
            ((__bf16*)Cout)[(size_t)col * M + row] = (__bf16)v;
          else
            ((__bf16*)Cout)[(size_t)row * N + col] = (__bf16)v;
        } else {
          ((float*)Cout)[(size_t)row * N + col] = v;
        }
      }
    }
  }
}

// ---- Flash attention (GQA) with TDM double-buffered K/V^T tiles ------------
// grid = (S/64, NUM_HEADS), block = 128 (4 waves). Wave w owns 16 query rows.
__global__ void __launch_bounds__(128)
gqa_attn(const __bf16* __restrict__ Q, const __bf16* __restrict__ Kb,
         const __bf16* __restrict__ VT, __bf16* __restrict__ X) {
  __shared__ __align__(16) __bf16 sK[2][64 * 128];   // [key][d]
  __shared__ __align__(16) __bf16 sVT[2][128 * 64];  // [d][key]
  __shared__ __align__(16) __bf16 sP[4][16 * 64];    // per-wave probs

  const int t    = threadIdx.x;
  const int lane = t & 31;
  const int w    = t >> 5;
  const int l    = lane & 15;
  const int hi   = (lane >> 4) & 1;
  const int h    = blockIdx.y;
  const int g    = h >> 2;
  const int q0   = blockIdx.x * 64 + w * 16;

  v16bf qf[4];
#pragma unroll
  for (int kc = 0; kc < 4; ++kc)
    qf[kc] =
        ldfragA(Q + (size_t)(q0 + l) * HIDDEN + h * HEAD_DIM + kc * 32 + hi * 16);

  v8f of[8];
#pragma unroll
  for (int d8 = 0; d8 < 8; ++d8) of[d8] = (v8f)0.0f;
  float mrow[8], lrow[8];
#pragma unroll
  for (int r = 0; r < 8; ++r) { mrow[r] = -1e30f; lrow[r] = 0.0f; }

  const float scale = 0.08838834764831845f;  // 1/sqrt(128)
  const int njt = S_LEN / 64;

  // Prime tile pair 0: K tile [64 x 128] and V^T tile [128 x 64] via TDM.
  if (t < 32) {
    tdm_load_2d(lds_off(sK[0]), Kb + /*j0=0*/ (size_t)0 * KV_DIM + g * HEAD_DIM,
                128, 64, KV_DIM);
    tdm_load_2d(lds_off(sVT[0]), VT + (size_t)(g * HEAD_DIM) * S_LEN + 0,
                64, 128, S_LEN);
  }

  for (int jt = 0; jt < njt; ++jt) {
    const int cur = jt & 1;
    if (t < 32) {
      if (jt + 1 < njt) {
        const int j0n = (jt + 1) * 64;
        tdm_load_2d(lds_off(sK[1 - cur]),
                    Kb + (size_t)j0n * KV_DIM + g * HEAD_DIM, 128, 64, KV_DIM);
        tdm_load_2d(lds_off(sVT[1 - cur]),
                    VT + (size_t)(g * HEAD_DIM) * S_LEN + j0n, 64, 128, S_LEN);
        __builtin_amdgcn_s_wait_tensorcnt(2);  // current tile pair complete
      } else {
        __builtin_amdgcn_s_wait_tensorcnt(0);
      }
    }
    __syncthreads();

    const __bf16* kTile = sK[cur];
    const __bf16* vTile = sVT[cur];

    // S = Q @ K^T (16x64 stripe per wave)
    v8f st[4];
#pragma unroll
    for (int ct = 0; ct < 4; ++ct) {
      st[ct] = (v8f)0.0f;
#pragma unroll
      for (int kc = 0; kc < 4; ++kc)
        st[ct] = wmma_bf16(
            qf[kc], ldfragB(kTile + (ct * 16 + l) * 128 + kc * 32 + hi * 16),
            st[ct]);
    }

    // Online softmax: row stats via 16-lane shuffles
    float alpha[8], mnew[8];
#pragma unroll
    for (int r = 0; r < 8; ++r) {
      float mx = -1e30f;
#pragma unroll
      for (int ct = 0; ct < 4; ++ct) {
        st[ct][r] *= scale;
        mx = fmaxf(mx, st[ct][r]);
      }
#pragma unroll
      for (int off = 8; off >= 1; off >>= 1)
        mx = fmaxf(mx, __shfl_xor(mx, off, 32));
      mnew[r]  = fmaxf(mrow[r], mx);
      alpha[r] = __expf(mrow[r] - mnew[r]);
      mrow[r]  = mnew[r];
    }
#pragma unroll
    for (int r = 0; r < 8; ++r) {
      float s = 0.0f;
#pragma unroll
      for (int ct = 0; ct < 4; ++ct) {
        float p = __expf(st[ct][r] - mnew[r]);
        st[ct][r] = p;
        s += p;
      }
#pragma unroll
      for (int off = 8; off >= 1; off >>= 1) s += __shfl_xor(s, off, 32);
      lrow[r] = lrow[r] * alpha[r] + s;
    }
#pragma unroll
    for (int d8 = 0; d8 < 8; ++d8)
#pragma unroll
      for (int r = 0; r < 8; ++r) of[d8][r] *= alpha[r];

    // Stage P as bf16 in A layout source (wave-private region; LDS in-order)
    __bf16* pw = &sP[w][0];
#pragma unroll
    for (int ct = 0; ct < 4; ++ct)
#pragma unroll
      for (int r = 0; r < 8; ++r)
        pw[(hi * 8 + r) * 64 + ct * 16 + l] = (__bf16)st[ct][r];

    // O += P @ V
    v16bf pa[2];
#pragma unroll
    for (int kc = 0; kc < 2; ++kc)
      pa[kc] = ldfragA(pw + l * 64 + kc * 32 + hi * 16);
#pragma unroll
    for (int d8 = 0; d8 < 8; ++d8)
#pragma unroll
      for (int kc = 0; kc < 2; ++kc)
        of[d8] = wmma_bf16(
            pa[kc], ldfragB(vTile + (d8 * 16 + l) * 64 + kc * 32 + hi * 16),
            of[d8]);
    __syncthreads();  // done reading tile pair before TDM refills it
  }

  // Normalize; store in reference's swapaxes(-1,-2).reshape flat order
#pragma unroll
  for (int d8 = 0; d8 < 8; ++d8) {
#pragma unroll
    for (int r = 0; r < 8; ++r) {
      int s = q0 + hi * 8 + r;
      int d = d8 * 16 + l;
      float v = of[d8][r] / lrow[r];
      size_t flat = (size_t)h * (HEAD_DIM * S_LEN) + (size_t)d * S_LEN + s;
      X[flat] = (__bf16)v;
    }
  }
}

// ---------------------------------------------------------------------------
extern "C" void kernel_launch(void* const* d_in, const int* in_sizes, int n_in,
                              void* d_out, int out_size, void* d_ws,
                              size_t ws_size, hipStream_t stream) {
  const float* hidden = (const float*)d_in[0];
  const float* wq     = (const float*)d_in[1];
  const float* bq     = (const float*)d_in[2];
  const float* wk     = (const float*)d_in[3];
  const float* bk     = (const float*)d_in[4];
  const float* wv     = (const float*)d_in[5];
  const float* bv     = (const float*)d_in[6];
  const float* wo     = (const float*)d_in[7];
  const float* bo     = (const float*)d_in[8];
  (void)in_sizes; (void)n_in; (void)out_size; (void)ws_size;

  size_t off = 0;
  auto take = [&](size_t elems) {
    void* p = (char*)d_ws + off;
    off += (elems * sizeof(__bf16) + 255) & ~(size_t)255;
    return (__bf16*)p;
  };
  __bf16* hbf  = take((size_t)S_LEN * HIDDEN);
  __bf16* wqT  = take((size_t)HIDDEN * HIDDEN);  // [N][K]
  __bf16* wkT  = take((size_t)KV_DIM * HIDDEN);
  __bf16* wvT  = take((size_t)KV_DIM * HIDDEN);
  __bf16* woT  = take((size_t)HIDDEN * HIDDEN);
  __bf16* qb   = take((size_t)S_LEN * HIDDEN);   // [S][H]
  __bf16* kb   = take((size_t)S_LEN * KV_DIM);   // [S][KV]
  __bf16* vT   = take((size_t)KV_DIM * S_LEN);   // [KV][S]
  __bf16* xb   = take((size_t)S_LEN * HIDDEN);   // scrambled attn out

  cvt_f32_bf16<<<4096, 256, 0, stream>>>(hidden, hbf, S_LEN * HIDDEN);

  // Weights: f32 [K][N] -> bf16 transposed [N][K]
  transpose_cvt<<<dim3(HIDDEN / 32, HIDDEN / 32), 256, 0, stream>>>(
      wq, wqT, HIDDEN, HIDDEN);
  transpose_cvt<<<dim3(KV_DIM / 32, HIDDEN / 32), 256, 0, stream>>>(
      wk, wkT, HIDDEN, KV_DIM);
  transpose_cvt<<<dim3(KV_DIM / 32, HIDDEN / 32), 256, 0, stream>>>(
      wv, wvT, HIDDEN, KV_DIM);
  transpose_cvt<<<dim3(HIDDEN / 32, HIDDEN / 32), 256, 0, stream>>>(
      wo, woT, HIDDEN, HIDDEN);

  // Projections (bf16 out). V is written transposed for attention staging.
  gemm_bias<true, false><<<dim3(HIDDEN / 128, S_LEN / 128), 256, 0, stream>>>(
      hbf, wqT, bq, qb, S_LEN, HIDDEN, HIDDEN);
  gemm_bias<true, false><<<dim3(KV_DIM / 128, S_LEN / 128), 256, 0, stream>>>(
      hbf, wkT, bk, kb, S_LEN, KV_DIM, HIDDEN);
  gemm_bias<true, true><<<dim3(KV_DIM / 128, S_LEN / 128), 256, 0, stream>>>(
      hbf, wvT, bv, vT, S_LEN, KV_DIM, HIDDEN);

  gqa_attn<<<dim3(S_LEN / 64, NUM_HEADS), 128, 0, stream>>>(qb, kb, vT, xb);

  // Output projection (f32 out to d_out)
  gemm_bias<false, false><<<dim3(HIDDEN / 128, S_LEN / 128), 256, 0, stream>>>(
      xb, woT, bo, d_out, S_LEN, HIDDEN, HIDDEN);
}
